// DeformConvBNRelu_36189394436411
// MI455X (gfx1250) — matrix operand order repaired and checked by reading, not compiled, not run on hardware
//
#include <hip/hip_runtime.h>
#include <hip/hip_bf16.h>
#include <math.h>

#define B_    16
#define CIN   128
#define HH    64
#define WW    64
#define COUT  256
#define NPIX  (HH*WW)          // 4096
#define KDIM  (CIN*9)          // 1152
#define KSTEPS (KDIM/32)       // 36
#define EPS_  1e-5f

typedef __attribute__((ext_vector_type(16))) __bf16 v16bf;
typedef __attribute__((ext_vector_type(8)))  float  v8f;
typedef __attribute__((ext_vector_type(4)))  unsigned int v4u;

// workspace layout (byte offsets)
//   off       : 0         .. +4,718,592   (16*18*4096 f32, offset-conv output)
//   wpack     : 4,718,592 .. +589,824     (36*16*512 bf16 main-weight B-fragments)
//   wpack_off : 5,308,416 .. +73,728      (36*2*512 bf16 offset-weight B-fragments, N pad 18->32)
//   stats     : 5,382,144 .. +4,096       (sum[256], sumsq[256], scale[256], shift[256])
//   ybuf      : 8,388,608 .. +67,108,864  (raw conv output, f32 NCHW)
#define WS_OFF_OFF    0
#define WS_WPACK_OFF  4718592
#define WS_WPOFF_OFF  5308416
#define WS_STATS_OFF  5382144
#define WS_Y_OFF      8388608

__device__ __forceinline__ unsigned short f2bf(float f) {
    unsigned int u = __float_as_uint(f);
    u += 0x7FFFu + ((u >> 16) & 1u);   // round-to-nearest-even
    return (unsigned short)(u >> 16);
}

// ---- Kernel P1: pack main weight into WMMA B-fragment layout, zero stats ----
// K ordering: k = tap*128 + c.  Block (ks,nt): 32 lanes x 16 bf16.
// lanes 0-15: N=lane,    K = ks*32 + j      (j=0..15)
// lanes16-31: N=lane-16, K = ks*32 + 16 + j
__global__ __launch_bounds__(256) void k_pack(
    const float* __restrict__ w, unsigned short* __restrict__ wpack,
    float* __restrict__ stats) {
    int t = blockIdx.x * 256 + threadIdx.x;           // 36*16*512 = 294912
    if (t < 1024) stats[t] = 0.0f;
    int j    = t & 15;
    int lane = (t >> 4) & 31;
    int nt   = (t >> 9) & 15;
    int ks   = t >> 13;
    int n    = nt * 16 + (lane & 15);
    int k    = ks * 32 + ((lane < 16) ? j : 16 + j);
    int tap  = k >> 7;
    int c    = k & 127;
    wpack[t] = f2bf(w[((size_t)n * CIN + c) * 9 + tap]);
}

// ---- Kernel P2: pack offset-conv weight (N padded 18 -> 32) ----
__global__ __launch_bounds__(256) void k_pack_off(
    const float* __restrict__ w_off, unsigned short* __restrict__ wpo) {
    int t = blockIdx.x * 256 + threadIdx.x;           // 36*2*512 = 36864
    int j    = t & 15;
    int lane = (t >> 4) & 31;
    int nt   = (t >> 9) & 1;
    int ks   = t >> 10;
    int n    = nt * 16 + (lane & 15);
    int k    = ks * 32 + ((lane < 16) ? j : 16 + j);
    int tap  = k >> 7;
    int c    = k & 127;
    wpo[t] = (n < 18) ? f2bf(w_off[((size_t)n * CIN + c) * 9 + tap])
                      : (unsigned short)0;
}

// ---- Kernel 1: offset conv as WMMA implicit GEMM (M=64px, N=32(18), K=1152) ----
__global__ __launch_bounds__(256) void k_offset_wmma(
    const float* __restrict__ x, const unsigned short* __restrict__ wpo,
    const float* __restrict__ b_off, float* __restrict__ off) {
    __shared__ __align__(16) unsigned short sCols[64 * 1160];   // im2col, bf16

    const int tile = blockIdx.x;          // b*64 + yrow
    const int b    = tile >> 6;
    const int yrow = tile & 63;
    const int tid  = threadIdx.x;
    const int lane = tid & 31;
    const int wave = tid >> 5;
    const int half = (lane >> 4) & 1;

    // Phase 1: regular im2col (integer taps, zero-pad) -> LDS bf16
    const float* xw0 = x + ((size_t)b * CIN + wave * 16) * NPIX;
    for (int tap = 0; tap < 9; tap++) {
        const int kh = tap / 3, kw = tap % 3;
        const int yy = yrow + kh - 1;
        const bool vy = (yy >= 0) && (yy < HH);
        const int cy = min(max(yy, 0), HH - 1);
        #pragma unroll
        for (int p = 0; p < 2; p++) {
            const int m  = lane + p * 32;
            const int xx = m + kw - 1;
            const bool vv = vy && (xx >= 0) && (xx < WW);
            const int aoff = cy * WW + min(max(xx, 0), WW - 1);
            unsigned short* dst = &sCols[(size_t)m * 1160 + tap * 128 + wave * 16];
            #pragma unroll
            for (int cc = 0; cc < 16; cc += 2) {
                float v0 = vv ? xw0[cc * NPIX + aoff]       : 0.f;
                float v1 = vv ? xw0[(cc + 1) * NPIX + aoff] : 0.f;
                *(unsigned int*)&dst[cc] =
                    (unsigned)f2bf(v0) | ((unsigned)f2bf(v1) << 16);
            }
        }
    }
    __syncthreads();

    // Phase 2: one 16x16 tile per wave (mt = wave&3, nt = wave>>2)
    const int mt = wave & 3, nt = wave >> 2;
    union AB { v16bf v; v4u u[2]; };
    v8f acc = {0.f, 0.f, 0.f, 0.f, 0.f, 0.f, 0.f, 0.f};
    const int arow_k = half * 16;
    #pragma unroll 1
    for (int ks = 0; ks < KSTEPS; ks++) {
        AB a, bf;
        const unsigned short* ap =
            &sCols[(size_t)(mt * 16 + (lane & 15)) * 1160 + ks * 32 + arow_k];
        a.u[0] = *(const v4u*)(ap);
        a.u[1] = *(const v4u*)(ap + 8);
        const v4u* bp = (const v4u*)(wpo + ((size_t)(ks * 2 + nt) * 32 + lane) * 16);
        bf.u[0] = bp[0];
        bf.u[1] = bp[1];
        acc = __builtin_amdgcn_wmma_f32_16x16x32_bf16(
            false, a.v, false, bf.v, (short)0, acc, false, false);
    }

    // Phase 3: store valid channels (n < 18) + bias
    const int n = nt * 16 + (lane & 15);
    if (n < 18) {
        const float bias = b_off[n];
        #pragma unroll
        for (int v = 0; v < 8; v++) {
            int m = mt * 16 + half * 8 + v;
            off[(((size_t)b * 18 + n) * HH + yrow) * WW + m] = acc[v] + bias;
        }
    }
}

// ---- Kernel 2: fused bilinear sampling + WMMA GEMM + stats + y store ----
__global__ __launch_bounds__(256) void k_deform_gemm(
    const float* __restrict__ x, const float* __restrict__ off,
    const unsigned short* __restrict__ wpack,
    float* __restrict__ ybuf, float* __restrict__ stats) {
    __shared__ __align__(16) unsigned short sCols[64 * 1160];   // 148,480 B LDS

    const int tile = blockIdx.x;          // b*64 + yrow
    const int b    = tile >> 6;
    const int yrow = tile & 63;
    const int tid  = threadIdx.x;
    const int lane = tid & 31;
    const int wave = tid >> 5;
    const int half = (lane >> 4) & 1;

    // Phase 1: bilinear sample 64px x 9taps x 128ch into LDS (bf16)
    const float* xb   = x + (size_t)b * CIN * NPIX;
    const float* offb = off + (size_t)b * 18 * NPIX + (size_t)yrow * WW;
    const float* xw0  = xb + (size_t)(wave * 16) * NPIX;
    for (int tap = 0; tap < 9; tap++) {
        const int kh = tap / 3, kw = tap % 3;
        #pragma unroll
        for (int p = 0; p < 2; p++) {
            const int m = lane + p * 32;                  // pixel (x column)
            float dy = offb[(2 * tap) * NPIX + m];
            float dx = offb[(2 * tap + 1) * NPIX + m];
            float py = (float)(yrow - 1 + kh) + dy;
            float px = (float)(m - 1 + kw) + dx;
            float y0f = floorf(py), x0f = floorf(px);
            float fy = py - y0f,    fx = px - x0f;
            int y0 = (int)y0f, x0 = (int)x0f;
            int y1 = y0 + 1,   x1 = x0 + 1;
            bool vy0 = (y0 >= 0) & (y0 < HH), vy1 = (y1 >= 0) & (y1 < HH);
            bool vx0 = (x0 >= 0) & (x0 < WW), vx1 = (x1 >= 0) & (x1 < WW);
            int cy0 = min(max(y0, 0), HH - 1), cy1 = min(max(y1, 0), HH - 1);
            int cx0 = min(max(x0, 0), WW - 1), cx1 = min(max(x1, 0), WW - 1);
            float w00 = (vy0 && vx0) ? (1.f - fy) * (1.f - fx) : 0.f;
            float w01 = (vy0 && vx1) ? (1.f - fy) * fx         : 0.f;
            float w10 = (vy1 && vx0) ? fy * (1.f - fx)         : 0.f;
            float w11 = (vy1 && vx1) ? fy * fx                 : 0.f;
            int a00 = cy0 * WW + cx0, a01 = cy0 * WW + cx1;
            int a10 = cy1 * WW + cx0, a11 = cy1 * WW + cx1;
            unsigned short* dst = &sCols[(size_t)m * 1160 + tap * 128 + wave * 16];
            #pragma unroll 2
            for (int cc = 0; cc < 16; cc += 2) {
                const float* p0 = xw0 + cc * NPIX;
                const float* p1 = xw0 + (cc + 1) * NPIX;
                float v0 = w00 * p0[a00] + w01 * p0[a01] + w10 * p0[a10] + w11 * p0[a11];
                float v1 = w00 * p1[a00] + w01 * p1[a01] + w10 * p1[a10] + w11 * p1[a11];
                *(unsigned int*)&dst[cc] =
                    (unsigned)f2bf(v0) | ((unsigned)f2bf(v1) << 16);
            }
        }
    }
    __syncthreads();

    // Phase 2: WMMA GEMM. Wave handles 64px x 32 output channels
    union AB { v16bf v; v4u u[2]; };
    v8f zacc = {0.f, 0.f, 0.f, 0.f, 0.f, 0.f, 0.f, 0.f};
    v8f acc[4][2];
    #pragma unroll
    for (int i = 0; i < 4; i++)
        #pragma unroll
        for (int j = 0; j < 2; j++) acc[i][j] = zacc;

    const int arow_k = half * 16;  // A-fragment K half-offset (contiguous, matches wpack)
    #pragma unroll 1
    for (int ks = 0; ks < KSTEPS; ks++) {
        AB a[4], bf[2];
        #pragma unroll
        for (int mt = 0; mt < 4; mt++) {
            const unsigned short* ap =
                &sCols[(size_t)(mt * 16 + (lane & 15)) * 1160 + ks * 32 + arow_k];
            a[mt].u[0] = *(const v4u*)(ap);        // K = ks*32 + half*16 + 0..7
            a[mt].u[1] = *(const v4u*)(ap + 8);    // K = ks*32 + half*16 + 8..15
        }
        #pragma unroll
        for (int nt2 = 0; nt2 < 2; nt2++) {
            const v4u* bp = (const v4u*)(wpack +
                ((size_t)(ks * 16 + wave * 2 + nt2) * 32 + lane) * 16);
            bf[nt2].u[0] = bp[0];
            bf[nt2].u[1] = bp[1];
        }
        #pragma unroll
        for (int mt = 0; mt < 4; mt++)
            #pragma unroll
            for (int nt2 = 0; nt2 < 2; nt2++)
                acc[mt][nt2] = __builtin_amdgcn_wmma_f32_16x16x32_bf16(
                    false, a[mt].v, false, bf[nt2].v,
                    (short)0, acc[mt][nt2], false, false);
    }

    // Phase 3: per-channel stats + coalesced store via LDS transpose
    #pragma unroll
    for (int nt2 = 0; nt2 < 2; nt2++) {
        int n = (wave * 2 + nt2) * 16 + (lane & 15);
        float s = 0.f, sq = 0.f;
        #pragma unroll
        for (int mt = 0; mt < 4; mt++)
            #pragma unroll
            for (int v = 0; v < 8; v++) {
                float f = acc[mt][nt2][v];
                s += f; sq += f * f;
            }
        atomicAdd(&stats[n], s);
        atomicAdd(&stats[256 + n], sq);
    }
    __syncthreads();                       // done reading sCols -> safe to reuse
    float* tb = (float*)sCols;             // 256ch x 64px f32 = 64 KB
    #pragma unroll
    for (int nt2 = 0; nt2 < 2; nt2++) {
        int ch = wave * 32 + nt2 * 16 + (lane & 15);
        #pragma unroll
        for (int mt = 0; mt < 4; mt++)
            #pragma unroll
            for (int v = 0; v < 8; v++)
                tb[ch * 64 + mt * 16 + half * 8 + v] = acc[mt][nt2][v];
    }
    __syncthreads();
    #pragma unroll 4
    for (int it = 0; it < 16; it++) {
        int idx4 = it * 256 + tid;         // 4096 float4 elements
        int ch   = idx4 >> 4;
        int xo   = (idx4 & 15) << 2;
        float4 vv = *(const float4*)&tb[ch * 64 + xo];
        *(float4*)&ybuf[(((size_t)b * COUT + ch) * HH + yrow) * WW + xo] = vv;
    }
}

// ---- Kernel 3: finalize BN scale/shift ----
__global__ void k_finalize(const float* __restrict__ gamma,
                           const float* __restrict__ beta, float* stats) {
    int n = threadIdx.x;
    float cnt  = (float)(B_ * NPIX);
    float mean = stats[n] / cnt;
    float var  = stats[256 + n] / cnt - mean * mean;
    float sc   = gamma[n] * rsqrtf(var + EPS_);
    stats[512 + n] = sc;
    stats[768 + n] = beta[n] - mean * sc;
}

// ---- Kernel 4: apply BN + ReLU (streaming) ----
__global__ __launch_bounds__(256) void k_bnrelu(
    const float* __restrict__ ybuf, const float* __restrict__ stats,
    float* __restrict__ out) {
    size_t i = (size_t)blockIdx.x * 256 + threadIdx.x;  // over 4,194,304 float4
    int ch = (int)((i >> 10) & 255);
    float sc = stats[512 + ch], sh = stats[768 + ch];
    float4 v = ((const float4*)ybuf)[i];
    float4 r;
    r.x = fmaxf(fmaf(v.x, sc, sh), 0.f);
    r.y = fmaxf(fmaf(v.y, sc, sh), 0.f);
    r.z = fmaxf(fmaf(v.z, sc, sh), 0.f);
    r.w = fmaxf(fmaf(v.w, sc, sh), 0.f);
    ((float4*)out)[i] = r;
}

extern "C" void kernel_launch(void* const* d_in, const int* in_sizes, int n_in,
                              void* d_out, int out_size, void* d_ws, size_t ws_size,
                              hipStream_t stream) {
    const float* x      = (const float*)d_in[0];
    const float* w_off  = (const float*)d_in[1];
    const float* b_off  = (const float*)d_in[2];
    const float* weight = (const float*)d_in[3];
    const float* gamma  = (const float*)d_in[4];
    const float* beta   = (const float*)d_in[5];
    float* out = (float*)d_out;
    char*  ws  = (char*)d_ws;

    float*          off   = (float*)(ws + WS_OFF_OFF);
    unsigned short* wpack = (unsigned short*)(ws + WS_WPACK_OFF);
    unsigned short* wpo   = (unsigned short*)(ws + WS_WPOFF_OFF);
    float*          stats = (float*)(ws + WS_STATS_OFF);
    float*          ybuf  = (float*)(ws + WS_Y_OFF);

    k_pack<<<(KSTEPS * 16 * 512) / 256, 256, 0, stream>>>(weight, wpack, stats);
    k_pack_off<<<(KSTEPS * 2 * 512) / 256, 256, 0, stream>>>(w_off, wpo);
    k_offset_wmma<<<B_ * HH, 256, 0, stream>>>(x, wpo, b_off, off);
    k_deform_gemm<<<B_ * HH, 256, 0, stream>>>(x, off, wpack, ybuf, stats);
    k_finalize<<<1, 256, 0, stream>>>(gamma, beta, stats);
    k_bnrelu<<<(size_t)B_ * COUT * NPIX / 4 / 256, 256, 0, stream>>>(ybuf, stats, out);
}
